// MultiheadSelfAttention_40381282517486
// MI455X (gfx1250) — matrix operand order repaired and checked by reading, not compiled, not run on hardware
//
#include <hip/hip_runtime.h>
#include <hip/hip_bf16.h>
#include <hip/hip_fp16.h>

// MHSA forward for MI455X (gfx1250, wave32, WMMA).
// Pipeline: cast f32->f16, QKV GEMM (WMMA) with head-split scatter (+bias,
// +q-scale folded), flash attention (WMMA + async-to-LDS double-buffered K/V
// staging + online softmax in log2 domain), output projection GEMM (WMMA).

#define NB   2
#define NS   2048
#define NH   16
#define ND   64
#define NE   1024
#define NQKV 3072

typedef __attribute__((ext_vector_type(16))) _Float16 v16h;
typedef __attribute__((ext_vector_type(8)))  _Float16 v8h;
typedef __attribute__((ext_vector_type(4)))  _Float16 v4h;
typedef __attribute__((ext_vector_type(8)))  float    v8f;

// 0.125 * log2(e): folds the 1/sqrt(64) attention scale AND the exp->exp2
// conversion into Q so softmax can use raw v_exp_f32 (exp2) directly.
#define QSCALE 0.18033688011112042f

__device__ __forceinline__ v8f wmma16(v16h a, v16h b, v8f c) {
  return __builtin_amdgcn_wmma_f32_16x16x32_f16(
      /*neg_a=*/false, a, /*neg_b=*/false, b,
      /*c_mod=*/(short)0, c, /*reuse_a=*/false, /*reuse_b=*/false);
}

// A fragment: 16(M) x 32(K) f16, source row-major with K contiguous, leading dim ld.
// Lane L: row = L%16; K pairs {(L/16)*8..+7} and {16+(L/16)*8..+7}  (ISA 7.12.2).
__device__ __forceinline__ v16h load_fragA(const _Float16* base, int ld) {
  const int lane = threadIdx.x & 31;
  const _Float16* p = base + (lane & 15) * ld + (lane >> 4) * 8;
  union { v16h v; v8h h[2]; } u;
  u.h[0] = *(const v8h*)(p);
  u.h[1] = *(const v8h*)(p + 16);
  return u.v;
}

// B fragment: 32(K) x 16(N) f16, source N-major rows with K contiguous (i.e. B^T
// row-major), leading dim ld. Lane L: col N = L%16; K {(L/16)*8..+7} and {16+(L/16)*8..+7}.
__device__ __forceinline__ v16h load_fragB(const _Float16* base, int ld) {
  const int lane = threadIdx.x & 31;
  const _Float16* p = base + (lane & 15) * ld + (lane >> 4) * 8;
  union { v16h v; v8h h[2]; } u;
  u.h[0] = *(const v8h*)(p);
  u.h[1] = *(const v8h*)(p + 16);
  return u.v;
}

// gfx1250 async DMA: global -> LDS, 16 bytes per lane, tracked by ASYNCcnt.
// VDST = per-lane LDS byte offset VGPR, VADDR = per-lane 64-bit global address.
__device__ __forceinline__ void async_ld_b128(unsigned lds_off, const _Float16* g) {
  asm volatile("global_load_async_to_lds_b128 %0, %1, off"
               :: "v"(lds_off), "v"(g) : "memory");
}

// ---------------------------------------------------------------- cast f32->f16
__global__ void __launch_bounds__(256)
cast_f32_f16(const float* __restrict__ in, _Float16* __restrict__ out, int n) {
  int i = (blockIdx.x * 256 + threadIdx.x) * 4;
  if (i + 3 < n) {
    float4 f = *(const float4*)(in + i);
    v4h o = { (_Float16)f.x, (_Float16)f.y, (_Float16)f.z, (_Float16)f.w };
    *(v4h*)(out + i) = o;
  }
}

// --------------------------------------------------------------------- QKV GEMM
// C[4096,3072] = xh[4096,1024] @ wh[3072,1024]^T + bias, scattered:
//   e -> head h=e/192, t=(e%192)/64 (0=q,1=k,2=v), d=e%64.
//   Q,K stored [B,H,S,64]; V stored transposed [B,H,64,S]. Q scaled by QSCALE.
// Each wave: 64x64 register tile (16 WMMAs / k-step); 4 waves = 128x128 block.
__global__ void __launch_bounds__(128)
qkv_gemm(const _Float16* __restrict__ xh, const _Float16* __restrict__ wh,
         const float* __restrict__ bias,
         _Float16* __restrict__ qb, _Float16* __restrict__ kb,
         _Float16* __restrict__ vb) {
  const int lane = threadIdx.x & 31;
  const int wave = threadIdx.x >> 5;
  const int hl   = lane >> 4;
  const int m0 = blockIdx.y * 128 + (wave >> 1) * 64;
  const int n0 = blockIdx.x * 128 + (wave & 1) * 64;

  v8f acc[4][4] = {};
  for (int k0 = 0; k0 < NE; k0 += 32) {
    v16h a[4], b[4];
#pragma unroll
    for (int i = 0; i < 4; ++i)
      a[i] = load_fragA(xh + (size_t)(m0 + i * 16) * NE + k0, NE);
#pragma unroll
    for (int j = 0; j < 4; ++j)
      b[j] = load_fragB(wh + (size_t)(n0 + j * 16) * NE + k0, NE);
#pragma unroll
    for (int i = 0; i < 4; ++i)
#pragma unroll
      for (int j = 0; j < 4; ++j)
        acc[i][j] = wmma16(a[i], b[j], acc[i][j]);
  }

#pragma unroll
  for (int j = 0; j < 4; ++j) {
    const int e = n0 + j * 16 + (lane & 15);
    const int h = e / 192, w = e % 192, t = w / 64, d = w % 64;
    const float bv = bias[e];
    const float sc = (t == 0) ? QSCALE : 1.0f;
#pragma unroll
    for (int i = 0; i < 4; ++i)
#pragma unroll
      for (int r = 0; r < 8; ++r) {
        const int gm = m0 + i * 16 + r + 8 * hl;       // row in [B*S]
        const int bb = gm >> 11, s = gm & 2047;
        const _Float16 val = (_Float16)((acc[i][j][r] + bv) * sc);
        if (t == 0)
          qb[(((size_t)(bb * NH + h)) * NS + s) * ND + d] = val;
        else if (t == 1)
          kb[(((size_t)(bb * NH + h)) * NS + s) * ND + d] = val;
        else
          vb[(((size_t)(bb * NH + h)) * ND + d) * NS + s] = val;
      }
  }
}

// --------------------------------------------------------------- flash attention
// Block = 8 waves covering 128 consecutive query rows of one (b,h); all waves
// share K/V tiles (64 keys) staged in LDS by async-to-LDS DMA, double-buffered.
// Per 64-key step per wave: 8 score WMMAs, log2-domain online softmax
// (4 shuffles/WMMA amortized), P bounced through LDS, 8 PV WMMAs.
__global__ void __launch_bounds__(256)
flash_attn(const _Float16* __restrict__ qb, const _Float16* __restrict__ kb,
           const _Float16* __restrict__ vb, _Float16* __restrict__ ob) {
  // 72-half (144B) padded rows keep every b128 16B-aligned.
  __shared__ __align__(16) _Float16 ldsK[2][64 * 72];   // [key][d]
  __shared__ __align__(16) _Float16 ldsV[2][64 * 72];   // [d][key]
  __shared__ __align__(16) _Float16 ldsP[8][16 * 72];   // per-wave P scratch

  const int tid  = threadIdx.x;
  const int lane = tid & 31;
  const int wave = tid >> 5;
  const int hl   = lane >> 4;

  const int bh   = blockIdx.x >> 4;        // b*16+h  (0..31)
  const int qblk = blockIdx.x & 15;        // 128-row query block
  const int qt   = qblk * 8 + wave;        // this wave's 16-row tile (0..127)

  const _Float16* qp = qb + ((size_t)bh * NS + qt * 16) * ND;
  const _Float16* kp = kb + (size_t)bh * NS * ND;
  const _Float16* vp = vb + (size_t)bh * ND * NS;
  _Float16* pt = &ldsP[wave][0];

  // Issue one 64-key K+V tile: 1024 b128 transfers, 4 per thread.
  auto issue_tile = [&](int kt, int buf) {
#pragma unroll
    for (int s = 0; s < 2; ++s) {
      const int i   = tid + s * 256;       // 0..511
      const int row = i >> 3;              // key (for K) / d (for V)
      const int c   = i & 7;               // 16B chunk within 128B row
      async_ld_b128((unsigned)(size_t)&ldsK[buf][row * 72 + c * 8],
                    kp + (size_t)(kt + row) * ND + c * 8);
      async_ld_b128((unsigned)(size_t)&ldsV[buf][row * 72 + c * 8],
                    vp + (size_t)row * NS + kt + c * 8);
    }
  };

  const v16h aq0 = load_fragA(qp +  0, ND);   // d 0..31
  const v16h aq1 = load_fragA(qp + 32, ND);   // d 32..63

  v8f acc[4] = {};
  float m[8], l[8];
#pragma unroll
  for (int r = 0; r < 8; ++r) { m[r] = -1e30f; l[r] = 0.0f; }

  issue_tile(0, 0);
  for (int step = 0; step < NS / 64; ++step) {
    const int cur = step & 1;
    const int kt  = step * 64;
    if (step + 1 < NS / 64) {
      issue_tile(kt + 64, cur ^ 1);
      asm volatile("s_wait_asynccnt 0x4" ::: "memory");  // current tile landed
    } else {
      asm volatile("s_wait_asynccnt 0x0" ::: "memory");
    }
    __syncthreads();                                     // all waves' DMA done

    const _Float16* Kt = &ldsK[cur][0];
    const _Float16* Vt = &ldsV[cur][0];

    // scores: 4 key sub-tiles x 2 d-chunks
    v8f s[4];
#pragma unroll
    for (int t = 0; t < 4; ++t) {
      v8f z = {};
      z = wmma16(aq0, load_fragB(Kt + (t * 16) * 72 +  0, 72), z);
      z = wmma16(aq1, load_fragB(Kt + (t * 16) * 72 + 32, 72), z);
      s[t] = z;
    }

    // online softmax (log2 domain), row stats live in each 16-lane half
#pragma unroll
    for (int r = 0; r < 8; ++r) {
      float t0 = fmaxf(fmaxf(s[0][r], s[1][r]), fmaxf(s[2][r], s[3][r]));
      t0 = fmaxf(t0, __shfl_xor(t0, 1));
      t0 = fmaxf(t0, __shfl_xor(t0, 2));
      t0 = fmaxf(t0, __shfl_xor(t0, 4));
      t0 = fmaxf(t0, __shfl_xor(t0, 8));
      const float mn = fmaxf(m[r], t0);
      const float f  = __builtin_amdgcn_exp2f(m[r] - mn);
      float p[4];
#pragma unroll
      for (int t = 0; t < 4; ++t) p[t] = __builtin_amdgcn_exp2f(s[t][r] - mn);
      float rs = (p[0] + p[1]) + (p[2] + p[3]);
      rs += __shfl_xor(rs, 1);
      rs += __shfl_xor(rs, 2);
      rs += __shfl_xor(rs, 4);
      rs += __shfl_xor(rs, 8);
      l[r] = l[r] * f + rs;
      m[r] = mn;
      acc[0][r] *= f; acc[1][r] *= f; acc[2][r] *= f; acc[3][r] *= f;
      const int row = r + 8 * hl;
#pragma unroll
      for (int t = 0; t < 4; ++t)
        pt[row * 72 + t * 16 + (lane & 15)] = (_Float16)p[t];
    }
    asm volatile("s_wait_dscnt 0x0" ::: "memory");  // P write->read, same wave
    const v16h ap0 = load_fragA(pt +  0, 72);       // P keys 0..31
    const v16h ap1 = load_fragA(pt + 32, 72);       // P keys 32..63
#pragma unroll
    for (int j = 0; j < 4; ++j) {
      acc[j] = wmma16(ap0, load_fragB(Vt + (j * 16) * 72 +  0, 72), acc[j]);
      acc[j] = wmma16(ap1, load_fragB(Vt + (j * 16) * 72 + 32, 72), acc[j]);
    }
    __syncthreads();                                // protect buffer reuse
  }

  const int b = bh >> 4, h = bh & 15;
#pragma unroll
  for (int r = 0; r < 8; ++r) {
    const float inv = 1.0f / l[r];
    const int srow = qt * 16 + r + 8 * hl;
    _Float16* op = ob + ((size_t)b * NS + srow) * NE + h * ND;
#pragma unroll
    for (int j = 0; j < 4; ++j)
      op[j * 16 + (lane & 15)] = (_Float16)(acc[j][r] * inv);
  }
}

// --------------------------------------------------------------- output proj
// out[4096,1024] (f32) = ah[4096,1024] @ owh[1024,1024]^T + o_b
__global__ void __launch_bounds__(128)
out_gemm(const _Float16* __restrict__ ah, const _Float16* __restrict__ wh,
         const float* __restrict__ bias, float* __restrict__ out) {
  const int lane = threadIdx.x & 31;
  const int wave = threadIdx.x >> 5;
  const int hl   = lane >> 4;
  const int m0 = blockIdx.y * 128 + (wave >> 1) * 64;
  const int n0 = blockIdx.x * 128 + (wave & 1) * 64;

  v8f acc[4][4] = {};
  for (int k0 = 0; k0 < NE; k0 += 32) {
    v16h a[4], b[4];
#pragma unroll
    for (int i = 0; i < 4; ++i)
      a[i] = load_fragA(ah + (size_t)(m0 + i * 16) * NE + k0, NE);
#pragma unroll
    for (int j = 0; j < 4; ++j)
      b[j] = load_fragB(wh + (size_t)(n0 + j * 16) * NE + k0, NE);
#pragma unroll
    for (int i = 0; i < 4; ++i)
#pragma unroll
      for (int j = 0; j < 4; ++j)
        acc[i][j] = wmma16(a[i], b[j], acc[i][j]);
  }

#pragma unroll
  for (int j = 0; j < 4; ++j) {
    const int col = n0 + j * 16 + (lane & 15);
    const float bv = bias[col];
#pragma unroll
    for (int i = 0; i < 4; ++i)
#pragma unroll
      for (int r = 0; r < 8; ++r) {
        const int gm = m0 + i * 16 + r + 8 * hl;
        out[(size_t)gm * NE + col] = acc[i][j][r] + bv;
      }
  }
}

// ------------------------------------------------------------------- launcher
extern "C" void kernel_launch(void* const* d_in, const int* in_sizes, int n_in,
                              void* d_out, int out_size, void* d_ws, size_t ws_size,
                              hipStream_t stream) {
  (void)in_sizes; (void)n_in; (void)out_size; (void)ws_size;
  const float* x     = (const float*)d_in[0];   // [2,2048,1024]
  const float* qkv_w = (const float*)d_in[1];   // [3072,1024]
  const float* qkv_b = (const float*)d_in[2];   // [3072]
  const float* o_w   = (const float*)d_in[3];   // [1024,1024]
  const float* o_b   = (const float*)d_in[4];   // [1024]
  float* out = (float*)d_out;

  _Float16* ws   = (_Float16*)d_ws;             // 48 MB of f16 scratch
  _Float16* xh   = ws;                          // 4M halfs
  _Float16* wqkv = xh   + (size_t)4 * 1024 * 1024;  // 3M
  _Float16* owh  = wqkv + (size_t)3 * 1024 * 1024;  // 1M
  _Float16* qbuf = owh  + (size_t)1 * 1024 * 1024;  // 4M [B,H,S,64]
  _Float16* kbuf = qbuf + (size_t)4 * 1024 * 1024;  // 4M [B,H,S,64]
  _Float16* vbuf = kbuf + (size_t)4 * 1024 * 1024;  // 4M [B,H,64,S]
  _Float16* vals = vbuf + (size_t)4 * 1024 * 1024;  // 4M [B,S,1024]

  cast_f32_f16<<<4096, 256, 0, stream>>>(x,     xh,   4194304);
  cast_f32_f16<<<3072, 256, 0, stream>>>(qkv_w, wqkv, 3145728);
  cast_f32_f16<<<1024, 256, 0, stream>>>(o_w,   owh,  1048576);

  qkv_gemm<<<dim3(NQKV / 128, (NB * NS) / 128), 128, 0, stream>>>(
      xh, wqkv, qkv_b, qbuf, kbuf, vbuf);

  flash_attn<<<NB * NH * (NS / 128), 256, 0, stream>>>(
      qbuf, kbuf, vbuf, vals);

  out_gemm<<<dim3(NE / 128, (NB * NS) / 128), 128, 0, stream>>>(
      vals, owh, o_b, out);
}
